// RoIHeads_53102975647949
// MI455X (gfx1250) — matrix operand order repaired and checked by reading, not compiled, not run on hardware
//
#include <hip/hip_runtime.h>
#include <hip/hip_bf16.h>

typedef __attribute__((ext_vector_type(16))) __bf16          v16bf;
typedef __attribute__((ext_vector_type(8)))  float           v8f;
typedef __attribute__((ext_vector_type(8)))  unsigned short  ush8;

// ---------- bf16 helpers ----------
__device__ inline unsigned short f2bf(float f) {
    union { float f; unsigned u; } v; v.f = f;
    unsigned r = v.u + 0x7FFFu + ((v.u >> 16) & 1u);
    return (unsigned short)(r >> 16);
}
__device__ inline float bf2f(unsigned short h) {
    union { unsigned u; float f; } v; v.u = ((unsigned)h) << 16;
    return v.f;
}
__device__ inline v16bf ld8x2(const unsigned short* p0, const unsigned short* p1) {
    union { ush8 h[2]; v16bf v; } u;
    u.h[0] = *(const ush8*)p0;
    u.h[1] = *(const ush8*)p1;
    return u.v;
}

// ---------- elementwise converts ----------
__global__ void f32_to_bf16_k(const float* __restrict__ s, unsigned short* __restrict__ d, int n) {
    int i = blockIdx.x * 256 + threadIdx.x;
    if (i < n) d[i] = f2bf(s[i]);
}

// mconv_w[4][256][256][3][3] -> [l][o][(ky*3+kx)*256 + i] bf16
__global__ void conv_w_rearrange_k(const float* __restrict__ s, unsigned short* __restrict__ d) {
    int i = blockIdx.x * 256 + threadIdx.x;
    if (i >= 4 * 256 * 2304) return;
    int k  = i % 2304;
    int o  = (i / 2304) % 256;
    int l  = i / (2304 * 256);
    int bk = k / 256, ci = k % 256;
    int ky = bk / 3,  kx = bk % 3;
    d[i] = f2bf(s[(((size_t)(l * 256 + o) * 256 + ci) * 3 + ky) * 3 + kx]);
}

// deconv_w[c][o][p][q] -> dwT[(p*2+q)][o][c] f32
__global__ void dw_rearrange_k(const float* __restrict__ s, float* __restrict__ d) {
    int i = blockIdx.x * 256 + threadIdx.x;
    if (i >= 4 * 256 * 256) return;
    int c  = i % 256;
    int o  = (i / 256) % 256;
    int pq = i / 65536;
    d[i] = s[(size_t)c * 1024 + o * 4 + pq];
}

// ---------- RoI align ----------
__device__ inline float bilinear_sample(const float* __restrict__ feat, int H, int b, int c,
                                        float y, float x) {
    bool valid = (y > -1.0f) && (y < (float)H) && (x > -1.0f) && (x < (float)H);
    float yc = fminf(fmaxf(y, 0.f), (float)H - 1.f);
    float xc = fminf(fmaxf(x, 0.f), (float)H - 1.f);
    int y0 = (int)fminf(fmaxf(floorf(yc), 0.f), (float)H - 2.f);
    int x0 = (int)fminf(fmaxf(floorf(xc), 0.f), (float)H - 2.f);
    float ly = yc - (float)y0, lx = xc - (float)x0;
    const float* p = feat + (((size_t)b * 256 + c) * H + y0) * H + x0;
    float f00 = p[0], f01 = p[1], f10 = p[H], f11 = p[H + 1];
    float v = f00 * (1.f - ly) * (1.f - lx) + f01 * (1.f - ly) * lx
            + f10 * ly * (1.f - lx) + f11 * ly * lx;
    return valid ? v : 0.f;
}

// mode 0: out_size=7,  out[r*12544 + c*49 + s]          (FC1 A-matrix layout)
// mode 1: out_size=14, out[(r*196 + s)*256 + c]         (conv input layout)
__global__ __launch_bounds__(256)
void roi_align_k(const float* __restrict__ f0, const float* __restrict__ f1,
                 const float* __restrict__ f2, const float* __restrict__ f3,
                 const float* __restrict__ boxes, const int* __restrict__ bidx,
                 unsigned short* __restrict__ out, int out_size, int mode) {
    int nss = out_size * out_size;
    int r  = blockIdx.x / nss;
    int s  = blockIdx.x % nss;
    int oy = s / out_size, ox = s % out_size;
    int c  = threadIdx.x;

    float x1 = boxes[r * 4 + 0], y1 = boxes[r * 4 + 1];
    float x2 = boxes[r * 4 + 2], y2 = boxes[r * 4 + 3];
    float area = fmaxf(x2 - x1, 0.f) * fmaxf(y2 - y1, 0.f);
    float lv = floorf(4.0f + log2f(sqrtf(area) / 224.0f + 1e-6f));
    lv = fminf(fmaxf(lv, 2.f), 5.f);
    int lvl = (int)lv - 2;

    const float* feats[4] = { f0, f1, f2, f3 };
    const int   HWs[4]    = { 200, 100, 50, 25 };
    const float scs[4]    = { 0.25f, 0.125f, 0.0625f, 0.03125f };
    const float* feat = feats[lvl];
    int H = HWs[lvl];
    float sc = scs[lvl];
    int b = bidx[r];

    float bx1 = x1 * sc, by1 = y1 * sc, bx2 = x2 * sc, by2 = y2 * sc;
    float rw = fmaxf(bx2 - bx1, 1.f), rh = fmaxf(by2 - by1, 1.f);
    float inv = 1.0f / (float)(2 * out_size);

    float acc = 0.f;
#pragma unroll
    for (int p = 0; p < 2; p++) {
        float gy = ((float)(2 * oy + p) + 0.5f) * inv;
        float yy = by1 + rh * gy;
#pragma unroll
        for (int q = 0; q < 2; q++) {
            float gx = ((float)(2 * ox + q) + 0.5f) * inv;
            float xx = bx1 + rw * gx;
            acc += bilinear_sample(feat, H, b, c, yy, xx);
        }
    }
    float v = acc * 0.25f;
    if (mode == 0) out[(size_t)r * 12544 + c * 49 + s] = f2bf(v);
    else           out[((size_t)r * 196 + s) * 256 + c] = f2bf(v);
}

// ---------- WMMA GEMM: C[M,N] = relu(A[M,K] @ B[N,K]^T + bias[N]) ----------
// A,B bf16 row-major. outMode 0: bf16 out, 1: f32 out. M,N mult of 32, K mult of 32.
__global__ __launch_bounds__(256)
void gemm_bf16_k(const unsigned short* __restrict__ A, const unsigned short* __restrict__ Bw,
                 const float* __restrict__ bias, void* __restrict__ Cout,
                 int M, int N, int K, int outMode) {
    int lane = threadIdx.x & 31;
    int wav  = threadIdx.x >> 5;
    int tile = blockIdx.x * 8 + wav;
    int tilesN = N >> 5;
    int mi = tile / tilesN, ni = tile % tilesN;
    int lm = lane & 15;
    bool hi = lane >= 16;

    int m0 = (mi << 5) + lm, m1 = m0 + 16;
    int n0 = (ni << 5) + lm, n1 = n0 + 16;
    const unsigned short* a0p = A  + (size_t)m0 * K;
    const unsigned short* a1p = A  + (size_t)m1 * K;
    const unsigned short* b0p = Bw + (size_t)n0 * K;
    const unsigned short* b1p = Bw + (size_t)n1 * K;
    int ka = hi ? 8 : 0;
    int kb = hi ? 16 : 0;

    v8f c00 = {0,0,0,0,0,0,0,0}, c01 = {0,0,0,0,0,0,0,0};
    v8f c10 = {0,0,0,0,0,0,0,0}, c11 = {0,0,0,0,0,0,0,0};

    for (int kt = 0; kt < K; kt += 32) {
        v16bf a0 = ld8x2(a0p + kt + ka, a0p + kt + ka + 16);
        v16bf a1 = ld8x2(a1p + kt + ka, a1p + kt + ka + 16);
        v16bf b0 = ld8x2(b0p + kt + ka, b0p + kt + ka + 16);
        v16bf b1 = ld8x2(b1p + kt + ka, b1p + kt + ka + 16);
        c00 = __builtin_amdgcn_wmma_f32_16x16x32_bf16(false, a0, false, b0, (short)0, c00, false, false);
        c01 = __builtin_amdgcn_wmma_f32_16x16x32_bf16(false, a0, false, b1, (short)0, c01, false, false);
        c10 = __builtin_amdgcn_wmma_f32_16x16x32_bf16(false, a1, false, b0, (short)0, c10, false, false);
        c11 = __builtin_amdgcn_wmma_f32_16x16x32_bf16(false, a1, false, b1, (short)0, c11, false, false);
    }

    int rb = hi ? 8 : 0;
    unsigned short* outh = (unsigned short*)Cout;
    float*          outf = (float*)Cout;
#pragma unroll
    for (int j = 0; j < 8; j++) {
        int r  = j + rb;
        int ms[2] = { (mi << 5) + r, (mi << 5) + 16 + r };
        int ns[2] = { (ni << 5) + lm, (ni << 5) + 16 + lm };
        float vals[2][2] = { { c00[j], c01[j] }, { c10[j], c11[j] } };
#pragma unroll
        for (int sm = 0; sm < 2; sm++)
#pragma unroll
            for (int sn = 0; sn < 2; sn++) {
                float v = fmaxf(vals[sm][sn] + bias[ns[sn]], 0.f);
                size_t o = (size_t)ms[sm] * N + ns[sn];
                if (outMode == 0) outh[o] = f2bf(v);
                else              outf[o] = v;
            }
    }
}

// ---------- implicit-GEMM 3x3 conv, 256->256 ch, 14x14, pad 1, relu ----------
// X,Y layout [roi][196][256] bf16. Wt layout [o][(ky*3+kx)*256 + i] bf16.
#define CONV_LDS_HALFS (16 * 16 * 256)
__global__ __launch_bounds__(512)
void conv3x3_wmma_k(const unsigned short* __restrict__ X, const unsigned short* __restrict__ Wt,
                    const float* __restrict__ bias, unsigned short* __restrict__ Y) {
    __shared__ unsigned short lds[CONV_LDS_HALFS];   // 128 KB padded input tile
    int n = blockIdx.x;
    int tid = threadIdx.x;

    uint4* l4 = (uint4*)lds;
    uint4 z; z.x = z.y = z.z = z.w = 0u;
    for (int i = tid; i < CONV_LDS_HALFS / 8; i += 512) l4[i] = z;
    __syncthreads();
    const uint4* src = (const uint4*)(X + (size_t)n * 196 * 256);
    for (int q = tid; q < 196 * 32; q += 512) {
        int s = q >> 5, cc = q & 31;
        int y = s / 14, x = s % 14;
        l4[((y + 1) * 16 + (x + 1)) * 32 + cc] = src[q];
    }
    __syncthreads();

    int wav = tid >> 5, lane = tid & 31, lm = lane & 15;
    bool hi = lane >= 16;
    int obase = wav << 4;                         // wave -> 16 output channels
    const unsigned short* wrow = Wt + (size_t)(obase + lm) * 2304;
    int ka = hi ? 8 : 0;
    int kb = hi ? 16 : 0;

    for (int nt = 0; nt < 13; nt++) {             // 13 tiles of 16 spatial cols
        int s  = nt * 16 + lm;
        int scl = s < 196 ? s : 195;
        int y = scl / 14, x = scl % 14;
        v8f acc = {0,0,0,0,0,0,0,0};
        for (int kt = 0; kt < 2304; kt += 32) {
            int k2  = kt + ka;
            int blk = k2 >> 8, i0 = k2 & 255;
            int ky = blk / 3, kx = blk % 3;
            v16bf bf = ld8x2(lds + (((y + ky) << 4) + (x + kx)) * 256 + i0, lds + (((y + ky) << 4) + (x + kx)) * 256 + i0 + 16);
            v16bf af = ld8x2(wrow + kt + ka, wrow + kt + ka + 16);
            acc = __builtin_amdgcn_wmma_f32_16x16x32_bf16(false, af, false, bf, (short)0, acc, false, false);
        }
        int rb = hi ? 8 : 0;
#pragma unroll
        for (int j = 0; j < 8; j++) {
            int o = obase + j + rb;
            if (s < 196) {
                float v = fmaxf(acc[j] + bias[o], 0.f);
                Y[((size_t)n * 196 + s) * 256 + o] = f2bf(v);
            }
        }
    }
}

// ---------- cls logits + softmax ----------
__global__ __launch_bounds__(128)
void cls_softmax_k(const float* __restrict__ x2, const float* __restrict__ cw,
                   const float* __restrict__ cb, float* __restrict__ scores) {
    __shared__ float lg[81];
    int row = blockIdx.x, tid = threadIdx.x;
    if (tid < 81) {
        const float* xr = x2 + (size_t)row * 1024;
        const float* wr = cw + (size_t)tid * 1024;
        float zv = cb[tid];
        for (int k = 0; k < 1024; k++) zv += xr[k] * wr[k];
        lg[tid] = zv;
    }
    __syncthreads();
    if (tid == 0) {
        float m = lg[0];
        for (int i = 1; i < 81; i++) m = fmaxf(m, lg[i]);
        float ssum = 0.f;
        for (int i = 0; i < 81; i++) { float e = expf(lg[i] - m); lg[i] = e; ssum += e; }
        float inv = 1.f / ssum;
        for (int i = 0; i < 81; i++) scores[(size_t)row * 81 + i] = lg[i] * inv;
    }
}

// ---------- bbox deltas + decode ----------
__global__ __launch_bounds__(256)
void bbox_decode_k(const float* __restrict__ x2, const float* __restrict__ bw,
                   const float* __restrict__ bb, const float* __restrict__ props,
                   float* __restrict__ boxesOut) {
    int t = blockIdx.x * 256 + threadIdx.x;     // exactly 1024*81 threads
    int row = t / 81, c = t % 81;
    const float* xr = x2 + (size_t)row * 1024;
    float d[4];
#pragma unroll
    for (int j = 0; j < 4; j++) {
        const float* wr = bw + (size_t)(c * 4 + j) * 1024;
        float zv = bb[c * 4 + j];
        for (int k = 0; k < 1024; k++) zv += xr[k] * wr[k];
        d[j] = zv;
    }
    float px1 = props[row * 4], py1 = props[row * 4 + 1];
    float px2 = props[row * 4 + 2], py2 = props[row * 4 + 3];
    float w = px2 - px1, h = py2 - py1;
    float cx = px1 + 0.5f * w, cy = py1 + 0.5f * h;
    const float LOGC = 4.135166556742356f;
    float dx = d[0] * 0.1f, dy = d[1] * 0.1f;
    float dw = fminf(d[2] * 0.2f, LOGC), dh = fminf(d[3] * 0.2f, LOGC);
    float pcx = dx * w + cx, pcy = dy * h + cy;
    float pw = expf(dw) * w, ph = expf(dh) * h;
    float o0 = fminf(fmaxf(pcx - 0.5f * pw, 0.f), 800.f);
    float o1 = fminf(fmaxf(pcy - 0.5f * ph, 0.f), 800.f);
    float o2 = fminf(fmaxf(pcx + 0.5f * pw, 0.f), 800.f);
    float o3 = fminf(fmaxf(pcy + 0.5f * ph, 0.f), 800.f);
    float* dst = boxesOut + (size_t)t * 4;
    dst[0] = o0; dst[1] = o1; dst[2] = o2; dst[3] = o3;
}

// ---------- mask roi selection ----------
__global__ void pick_labels_k(const float* __restrict__ scores, const float* __restrict__ boxes,
                              int* __restrict__ labels, float* __restrict__ mboxes,
                              int* __restrict__ mbidx) {
    int m = threadIdx.x;                         // 256 threads, 1 block
    int idx = m < 128 ? m : (m + 384);           // [0..127, 512..639]
    const float* sr = scores + (size_t)idx * 81;
    float best = sr[1]; int bc = 1;
    for (int c = 2; c < 81; c++) { float v = sr[c]; if (v > best) { best = v; bc = c; } }
    labels[m] = bc;
    const float* b = boxes + ((size_t)idx * 81 + bc) * 4;
#pragma unroll
    for (int j = 0; j < 4; j++) mboxes[m * 4 + j] = b[j];
    mbidx[m] = m < 128 ? 0 : 1;
}

// ---------- deconv (2x2 s2) + relu + per-label 1x1 + sigmoid ----------
__global__ __launch_bounds__(256)
void deconv_mask_k(const unsigned short* __restrict__ X4, const float* __restrict__ dwT,
                   const float* __restrict__ db, const float* __restrict__ mw,
                   const float* __restrict__ mb, const int* __restrict__ labels,
                   float* __restrict__ masks) {
    __shared__ float xs[256];
    __shared__ float red[256];
    int bi = blockIdx.x;                         // b*784 + h*28 + w
    int b = bi / 784;
    int hw = bi % 784;
    int h = hw / 28, w = hw % 28;
    int i = h >> 1, p = h & 1, j = w >> 1, q = w & 1;
    int s = i * 14 + j;
    int o = threadIdx.x;
    xs[o] = bf2f(X4[((size_t)b * 196 + s) * 256 + o]);
    __syncthreads();
    const float* wp = dwT + (size_t)((p * 2 + q) * 256 + o) * 256;
    float t = db[o];
    for (int c2 = 0; c2 < 256; c2++) t += xs[c2] * wp[c2];
    t = fmaxf(t, 0.f);
    int lab = labels[b];
    red[o] = t * mw[(size_t)lab * 256 + o];
    __syncthreads();
    for (int st = 128; st > 0; st >>= 1) {
        if (o < st) red[o] += red[o + st];
        __syncthreads();
    }
    if (o == 0) {
        float ml = red[0] + mb[lab];
        masks[bi] = 1.f / (1.f + expf(-ml));
    }
}

// ---------- host launch ----------
extern "C" void kernel_launch(void* const* d_in, const int* in_sizes, int n_in,
                              void* d_out, int out_size, void* d_ws, size_t ws_size,
                              hipStream_t stream) {
    const float* f0       = (const float*)d_in[0];
    const float* f1       = (const float*)d_in[1];
    const float* f2       = (const float*)d_in[2];
    const float* f3       = (const float*)d_in[3];
    const float* props    = (const float*)d_in[4];
    const float* fc1_w    = (const float*)d_in[5];
    const float* fc1_b    = (const float*)d_in[6];
    const float* fc2_w    = (const float*)d_in[7];
    const float* fc2_b    = (const float*)d_in[8];
    const float* cls_w    = (const float*)d_in[9];
    const float* cls_b    = (const float*)d_in[10];
    const float* bbox_w   = (const float*)d_in[11];
    const float* bbox_b   = (const float*)d_in[12];
    const float* mconv_w  = (const float*)d_in[13];
    const float* mconv_b  = (const float*)d_in[14];
    const float* deconv_w = (const float*)d_in[15];
    const float* deconv_b = (const float*)d_in[16];
    const float* mlogit_w = (const float*)d_in[17];
    const float* mlogit_b = (const float*)d_in[18];
    const int*   batch_idx= (const int*)d_in[19];

    float* scores   = (float*)d_out;
    float* boxesOut = scores + (size_t)1024 * 81;
    float* masksOut = boxesOut + (size_t)1024 * 81 * 4;

    // carve scratch
    char* ws = (char*)d_ws;
    size_t off = 0;
    auto carve = [&](size_t bytes) { char* p = ws + off; off += (bytes + 255) & ~(size_t)255; return p; };
    unsigned short* bfA   = (unsigned short*)carve((size_t)1024 * 12544 * 2);
    unsigned short* w1bf  = (unsigned short*)carve((size_t)1024 * 12544 * 2);
    unsigned short* w2bf  = (unsigned short*)carve((size_t)1024 * 1024 * 2);
    unsigned short* x1bf  = (unsigned short*)carve((size_t)1024 * 1024 * 2);
    float*          x2f   = (float*)carve((size_t)1024 * 1024 * 4);
    int*            labels= (int*)carve(256 * 4);
    float*          mbox  = (float*)carve(256 * 4 * 4);
    int*            mbidx = (int*)carve(256 * 4);
    unsigned short* cwbf  = (unsigned short*)carve((size_t)4 * 256 * 2304 * 2);
    unsigned short* Xa    = (unsigned short*)carve((size_t)256 * 196 * 256 * 2);
    unsigned short* Xb    = (unsigned short*)carve((size_t)256 * 196 * 256 * 2);
    float*          dwT   = (float*)carve((size_t)4 * 256 * 256 * 4);
    (void)ws_size; (void)in_sizes; (void)n_in; (void)out_size;

    // weight converts
    { int n = 1024 * 12544; f32_to_bf16_k<<<(n + 255) / 256, 256, 0, stream>>>(fc1_w, w1bf, n); }
    { int n = 1024 * 1024;  f32_to_bf16_k<<<(n + 255) / 256, 256, 0, stream>>>(fc2_w, w2bf, n); }
    { int n = 4 * 256 * 2304; conv_w_rearrange_k<<<(n + 255) / 256, 256, 0, stream>>>(mconv_w, cwbf); }
    { int n = 4 * 256 * 256;  dw_rearrange_k<<<(n + 255) / 256, 256, 0, stream>>>(deconv_w, dwT); }

    // box head
    roi_align_k<<<1024 * 49, 256, 0, stream>>>(f0, f1, f2, f3, props, batch_idx, bfA, 7, 0);
    gemm_bf16_k<<<128, 256, 0, stream>>>(bfA, w1bf, fc1_b, x1bf, 1024, 1024, 12544, 0);
    gemm_bf16_k<<<128, 256, 0, stream>>>(x1bf, w2bf, fc2_b, x2f, 1024, 1024, 1024, 1);
    cls_softmax_k<<<1024, 128, 0, stream>>>(x2f, cls_w, cls_b, scores);
    bbox_decode_k<<<(1024 * 81) / 256, 256, 0, stream>>>(x2f, bbox_w, bbox_b, props, boxesOut);

    // mask head
    pick_labels_k<<<1, 256, 0, stream>>>(scores, boxesOut, labels, mbox, mbidx);
    roi_align_k<<<256 * 196, 256, 0, stream>>>(f0, f1, f2, f3, mbox, mbidx, Xa, 14, 1);
    unsigned short* cin = Xa;
    unsigned short* cout = Xb;
    for (int l = 0; l < 4; l++) {
        conv3x3_wmma_k<<<256, 512, 0, stream>>>(cin, cwbf + (size_t)l * 256 * 2304,
                                                mconv_b + (size_t)l * 256, cout);
        unsigned short* t = cin; cin = cout; cout = t;
    }
    // after 4 layers result is in `cin` (== Xa)
    deconv_mask_k<<<256 * 784, 256, 0, stream>>>(cin, dwT, deconv_b, mlogit_w, mlogit_b,
                                                 labels, masksOut);
}
